// TaskSpecificAttention_50165218017856
// MI455X (gfx1250) — matrix operand order repaired and checked by reading, not compiled, not run on hardware
//
#include <hip/hip_runtime.h>
#include <hip/hip_bf16.h>

typedef __attribute__((ext_vector_type(16))) _Float16 v16h;
typedef __attribute__((ext_vector_type(8)))  _Float16 v8h;
typedef __attribute__((ext_vector_type(8)))  float    v8f;
typedef __attribute__((ext_vector_type(4)))  float    f4;

#define EMB   300
#define KP    320               // padded K (multiple of 32)
#define SEQ   512
#define BATCH 32
#define HEADS 6
#define HDIM  50
#define DPAD  64                // padded head dim
#define NLAY  4
#define MROWS (SEQ * BATCH)     // 16384
#define N3E   (3 * EMB)         // 900
#define WIF_ROWS 1024           // padded rows for Wi (grid.y*128 coverage)
#define WOF_ROWS 384            // padded rows for Wo

// ---------------- wave helpers (wave32) ----------------
__device__ inline float wave_max(float v) {
#pragma unroll
  for (int o = 16; o > 0; o >>= 1) v = fmaxf(v, __shfl_xor(v, o, 32));
  return v;
}
__device__ inline float wave_sum(float v) {
#pragma unroll
  for (int o = 16; o > 0; o >>= 1) v += __shfl_xor(v, o, 32);
  return v;
}
__device__ inline unsigned wang(unsigned x) {
  x = (x ^ 61u) ^ (x >> 16); x *= 9u; x ^= x >> 4;
  x *= 0x27d4eb2du; x ^= x >> 15; return x;
}
__device__ inline v16h cat8(v8h a, v8h b) {
  return __builtin_shufflevector(a, b, 0,1,2,3,4,5,6,7,8,9,10,11,12,13,14,15);
}

// ---- CDNA5 async global->LDS copy (16B) + wait (ISA 08_async_tensor) ----
__device__ inline void async_g2l_b128(unsigned lds_off, const void* gaddr) {
  asm volatile("global_load_async_to_lds_b128 %0, %1, off"
               :: "v"(lds_off), "v"(gaddr) : "memory");
}
__device__ inline void wait_async0() {
  asm volatile("s_wait_asynccnt 0x0" ::: "memory");
}

// ---------------- embedding ----------------
__global__ void tsa_embed_kernel(const int* __restrict__ tokens,
                                 const float* __restrict__ W_emb,
                                 const float* __restrict__ pos_emb,
                                 float* __restrict__ w_embed,
                                 float* __restrict__ h) {
  long idx = (long)blockIdx.x * 256 + threadIdx.x;   // over S*B*E
  if (idx >= (long)SEQ * BATCH * EMB) return;
  int e  = (int)(idx % EMB);
  long sb = idx / EMB;
  int b  = (int)(sb % BATCH);
  int s  = (int)(sb / BATCH);
  int tok = tokens[s * BATCH + b];
  float we = W_emb[(long)tok * EMB + e];
  w_embed[idx] = we;
  h[idx] = we + 2.0f * pos_emb[s * EMB + e];
}

// -------- f32 (rows x 300) -> f16 zero-padded (rows_dst x 320) --------
__global__ void tsa_cvt_pad_kernel(const float* __restrict__ src,
                                   _Float16* __restrict__ dst,
                                   int rows_src, long total /* rows_dst*KP */) {
  long idx = (long)blockIdx.x * 256 + threadIdx.x;
  if (idx >= total) return;
  int e = (int)(idx % KP);
  long r = idx / KP;
  float v = (e < EMB && r < rows_src) ? src[r * EMB + e] : 0.0f;
  dst[idx] = (_Float16)v;
}

// -------- qkv (S,B,900) f32 -> qf/kf (B,H,S,64) f16, vT (B,H,64,S) f16 --------
__global__ void tsa_cvt_qkv_kernel(const float* __restrict__ qkv,
                                   _Float16* __restrict__ qf,
                                   _Float16* __restrict__ kf,
                                   _Float16* __restrict__ vtf) {
  long idx = (long)blockIdx.x * 256 + threadIdx.x;   // B*H*S*64
  if (idx >= (long)BATCH * HEADS * SEQ * DPAD) return;
  int dd = (int)(idx & (DPAD - 1));
  int s  = (int)((idx >> 6) & (SEQ - 1));
  long bh = idx >> 15;                               // b*H + h
  int hh = (int)(bh % HEADS);
  int bb = (int)(bh / HEADS);
  float qv = 0.0f, kv = 0.0f, vv = 0.0f;
  if (dd < HDIM) {
    long base = ((long)s * BATCH + bb) * N3E + hh * HDIM + dd;
    qv = qkv[base];
    kv = qkv[base + EMB];
    vv = qkv[base + 2 * EMB];
  }
  qf[idx] = (_Float16)qv;
  kf[idx] = (_Float16)kv;
  vtf[(bh * DPAD + dd) * SEQ + s] = (_Float16)vv;
}

// ---------------- WMMA GEMM (f16 in, f32 out): C(M,N) = A @ Wf^T + bias ----
// A: (M, 320) f16; Wf: (Npad, 320) f16 zero-padded; 256 thr = 8 waves,
// block tile 16(M) x 128(N), wave tile 16x16. A strip staged via async copy.
__global__ void tsa_gemm_f16_kernel(const _Float16* __restrict__ A,
                                    const _Float16* __restrict__ Wf,
                                    const float* __restrict__ bias,
                                    float* __restrict__ C,
                                    int N) {
  __shared__ _Float16 aLds[16 * KP];                  // 10 KB
  const int tid  = threadIdx.x;
  const int lane = tid & 31;
  const int wave = tid >> 5;
  const bool low = lane < 16;
  const int rrow = lane & 15;
  const int m0 = blockIdx.x * 16;
  const int n  = blockIdx.y * 128 + wave * 16 + rrow; // < Npad by construction

  // async-stage 16x320 f16 A strip (10240 B = 640 x b128)
  {
    unsigned lbase = (unsigned)(unsigned long long)(void*)aLds;
    const char* gsrc = (const char*)(A + (long)m0 * KP);
    for (int t = tid; t < 640; t += 256)
      async_g2l_b128(lbase + t * 16, gsrc + t * 16);
    wait_async0();
  }
  __syncthreads();

  const int ab0 = low ? 0 : 8;     // A-fragment first contiguous run base
  const int bb0 = low ? 0 : 8;     // B-fragment first contiguous run base
  const _Float16* wrow = Wf + (long)n * KP;
  v8f acc = {};
#pragma unroll
  for (int kc = 0; kc < KP / 32; ++kc) {
    const int kk0 = kc * 32;
    v8h alo = *(const v8h*)(aLds + rrow * KP + kk0 + ab0);
    v8h ahi = *(const v8h*)(aLds + rrow * KP + kk0 + ab0 + 16);
    v8h blo = *(const v8h*)(wrow + kk0 + bb0);
    v8h bhi = *(const v8h*)(wrow + kk0 + bb0 + 16);
    acc = __builtin_amdgcn_wmma_f32_16x16x32_f16(false, cat8(alo, ahi),
                                                 false, cat8(blo, bhi),
                                                 (short)0, acc, false, false);
  }
  if (n < N) {
    float bv = bias[n];
#pragma unroll
    for (int r = 0; r < 8; ++r) {
      int row = low ? r : (r + 8);
      C[(long)(m0 + row) * N + n] = acc[r] + bv;
    }
  }
}

// ---------------- WMMA attention: one wave per (b, head, 16-query tile) ----
// qf/kf: (B,H,S,64) f16; vT: (B,H,64,S) f16; out: (S,B,300) f32.
__global__ void tsa_attention_kernel(const _Float16* __restrict__ qf,
                                     const _Float16* __restrict__ kf,
                                     const _Float16* __restrict__ vtf,
                                     float* __restrict__ attn_out) {
  extern __shared__ char smem[];
  _Float16* v_lds = (_Float16*)smem;                 // (64 x 512) f16 = 64KB
  float*    sc    = (float*)(smem + 64 * 1024);      // (16 x 512) f32 = 32KB
  const int lane = threadIdx.x & 31;
  const bool low = lane < 16;
  const int rrow = lane & 15;
  const int qt = blockIdx.x & 31;                    // S/16
  const long bh = blockIdx.x >> 5;                   // b*H + h
  const int hh = (int)(bh % HEADS);
  const int bb = (int)(bh / HEADS);
  const int q0 = qt * 16;
  const float scale = 0.14142135623730950f;          // 1/sqrt(50)
  const int ab0 = low ? 0 : 8;
  const int bb0 = low ? 0 : 8;

  // async-stage V^T slice (64 x 512 f16 = 64KB = 4096 x b128) into LDS
  {
    unsigned lbase = (unsigned)(unsigned long long)(void*)v_lds;
    const char* gsrc = (const char*)(vtf + bh * DPAD * SEQ);
    for (int t = threadIdx.x; t < 4096; t += 32)
      async_g2l_b128(lbase + t * 16, gsrc + t * 16);
    wait_async0();
  }

  // Q fragments: (16 x 64) -> two 16x32 chunks, all b128 loads
  const _Float16* qrow = qf + (bh * SEQ + q0 + rrow) * DPAD;
  v16h aq0 = cat8(*(const v8h*)(qrow + ab0),      *(const v8h*)(qrow + ab0 + 16));
  v16h aq1 = cat8(*(const v8h*)(qrow + 32 + ab0), *(const v8h*)(qrow + 32 + ab0 + 16));

  // scores = Q K^T * scale : 32 key tiles of 16
  for (int kt = 0; kt < 32; ++kt) {
    const _Float16* krow = kf + (bh * SEQ + kt * 16 + rrow) * DPAD;
    v16h bk0 = cat8(*(const v8h*)(krow + bb0),      *(const v8h*)(krow + bb0 + 16));
    v16h bk1 = cat8(*(const v8h*)(krow + 32 + bb0), *(const v8h*)(krow + 32 + bb0 + 16));
    v8f accs = {};
    accs = __builtin_amdgcn_wmma_f32_16x16x32_f16(false, aq0, false, bk0,
                                                  (short)0, accs, false, false);
    accs = __builtin_amdgcn_wmma_f32_16x16x32_f16(false, aq1, false, bk1,
                                                  (short)0, accs, false, false);
#pragma unroll
    for (int r = 0; r < 8; ++r) {
      int row = low ? r : (r + 8);
      sc[row * 512 + kt * 16 + rrow] = accs[r] * scale;
    }
  }
  __syncthreads();

  // per-row softmax across 512 keys (wave-parallel)
  for (int r = 0; r < 16; ++r) {
    float m = -1e30f;
    for (int c = lane; c < 512; c += 32) m = fmaxf(m, sc[r * 512 + c]);
    m = wave_max(m);
    float ssum = 0.0f;
    for (int c = lane; c < 512; c += 32) {
      float e = __expf(sc[r * 512 + c] - m);
      sc[r * 512 + c] = e;
      ssum += e;
    }
    ssum = wave_sum(ssum);
    float inv = 1.0f / ssum;
    for (int c = lane; c < 512; c += 32) sc[r * 512 + c] *= inv;
  }
  __syncthreads();

  // O = P @ V : accumulate 16 x 64 over 16 key-chunks of 32
  v8f acco[4] = {};
  for (int kc = 0; kc < 16; ++kc) {
    // P fragment: two contiguous 8-float runs from LDS, cvt to f16
    const float* prow = sc + rrow * 512 + kc * 32 + ab0;
    f4 x0 = *(const f4*)(prow);
    f4 x1 = *(const f4*)(prow + 4);
    f4 x2 = *(const f4*)(prow + 16);
    f4 x3 = *(const f4*)(prow + 20);
    v16h ap;
#pragma unroll
    for (int i = 0; i < 4; ++i) {
      ap[i]      = (_Float16)x0[i];
      ap[4 + i]  = (_Float16)x1[i];
      ap[8 + i]  = (_Float16)x2[i];
      ap[12 + i] = (_Float16)x3[i];
    }
#pragma unroll
    for (int nb = 0; nb < 4; ++nb) {
      const _Float16* vrow = v_lds + (nb * 16 + rrow) * 512 + kc * 32 + bb0;
      v16h bv = cat8(*(const v8h*)(vrow), *(const v8h*)(vrow + 16));
      acco[nb] = __builtin_amdgcn_wmma_f32_16x16x32_f16(false, ap, false, bv,
                                                        (short)0, acco[nb],
                                                        false, false);
    }
  }
#pragma unroll
  for (int nb = 0; nb < 4; ++nb) {
    int dd = nb * 16 + rrow;
    if (dd < HDIM) {
#pragma unroll
      for (int r = 0; r < 8; ++r) {
        int row = low ? r : (r + 8);
        attn_out[((long)(q0 + row) * BATCH + bb) * EMB + hh * HDIM + dd] = acco[nb][r];
      }
    }
  }
}

// ---------------- task attention (one block per batch, 512 threads = S) ----
__global__ void tsa_task_attn_kernel(const float* __restrict__ x,
                                     const float* __restrict__ te,
                                     unsigned seed,
                                     float* __restrict__ a_out) {
  __shared__ float red[512];
  __shared__ unsigned cnt;
  const int s = threadIdx.x;
  const int b = blockIdx.x;
  float w = 0.0f;
  const float* xr = x + ((long)s * BATCH + b) * EMB;
  for (int e = 0; e < EMB; ++e) w += xr[e] * te[e];
  red[s] = w;
  __syncthreads();
  for (int st = 256; st > 0; st >>= 1) {
    if (s < st) red[s] = fmaxf(red[s], red[s + st]);
    __syncthreads();
  }
  float mx = red[0];
  __syncthreads();
  unsigned hsh = wang(seed * 0x9E3779B1u ^ wang(b * 0x85EBCA6Bu + s * 0xC2B2AE35u));
  float u = (float)(hsh >> 8) * (1.0f / 16777216.0f) + 1e-7f;
  float g = -__logf(-__logf(u));
  float logit = (mx - w) + g;
  unsigned key = __float_as_uint(logit);
  key = (key & 0x80000000u) ? ~key : (key | 0x80000000u);
  unsigned prefix = 0u;
  for (int bit = 31; bit >= 0; --bit) {
    unsigned trial = prefix | (1u << bit);
    if (s == 0) cnt = 0u;
    __syncthreads();
    if (key >= trial) atomicAdd(&cnt, 1u);
    __syncthreads();
    if (cnt >= 256u) prefix = trial;
    __syncthreads();
  }
  float wm = (key >= prefix) ? -1e30f : w;
  red[s] = wm;
  __syncthreads();
  for (int st = 256; st > 0; st >>= 1) {
    if (s < st) red[s] = fmaxf(red[s], red[s + st]);
    __syncthreads();
  }
  float m2 = red[0];
  __syncthreads();
  float e = __expf(wm - m2);
  red[s] = e;
  __syncthreads();
  for (int st = 256; st > 0; st >>= 1) {
    if (s < st) red[s] += red[s + st];
    __syncthreads();
  }
  a_out[(long)b * SEQ + s] = e / red[0];
}

// ---------------- fused residual + LayerNorm (one wave per (s,b) row) ----
__global__ void tsa_fuse_ln_kernel(const float* __restrict__ proj,
                                   const float* __restrict__ w_embed,
                                   const float* __restrict__ h_old,
                                   const float* __restrict__ a1,
                                   const float* __restrict__ a2,
                                   const float* __restrict__ g,
                                   const float* __restrict__ bta,
                                   float* __restrict__ h_out) {
  const int wave = threadIdx.x >> 5;
  const int lane = threadIdx.x & 31;
  long row = (long)blockIdx.x * 4 + wave;
  if (row >= MROWS) return;
  int b = (int)(row % BATCH);
  int s = (int)(row / BATCH);
  float c1 = a1[b * SEQ + s];
  float c2 = a2[b * SEQ + s];
  float pre[10];
  float ssum = 0.0f, ssq = 0.0f;
#pragma unroll
  for (int t = 0; t < 10; ++t) {
    int e = lane + t * 32;
    float v = 0.0f;
    if (e < EMB) {
      long i = row * EMB + e;
      v = proj[i] + w_embed[i] * c1 + h_old[i] * c2;
    }
    pre[t] = v;
    ssum += v;
    ssq  += v * v;
  }
  ssum = wave_sum(ssum);
  ssq  = wave_sum(ssq);
  float mean = ssum * (1.0f / EMB);
  float var  = ssq * (1.0f / EMB) - mean * mean;
  float inv  = rsqrtf(var + 1e-12f);
#pragma unroll
  for (int t = 0; t < 10; ++t) {
    int e = lane + t * 32;
    if (e < EMB) h_out[row * EMB + e] = g[e] * (pre[t] - mean) * inv + bta[e];
  }
}

// ---------------- classifier head ----------------
__global__ void tsa_classifier_kernel(const float* __restrict__ h,
                                      const float* __restrict__ W_cls,
                                      const float* __restrict__ b_cls,
                                      float* __restrict__ out) {
  int t = threadIdx.x;
  if (t >= BATCH * 2) return;
  int b = t >> 1, c = t & 1;
  float acc = b_cls[c];
  const float* hr = h + (long)b * EMB;
  const float* wr = W_cls + (long)c * EMB;
  for (int e = 0; e < EMB; ++e) acc += hr[e] * wr[e];
  out[t] = 1.0f / (1.0f + __expf(-acc));
}

// ---------------- host orchestration ----------------
extern "C" void kernel_launch(void* const* d_in, const int* in_sizes, int n_in,
                              void* d_out, int out_size, void* d_ws, size_t ws_size,
                              hipStream_t stream) {
  const int*   tokens  = (const int*)d_in[0];
  const float* W_emb   = (const float*)d_in[1];
  const float* pos_emb = (const float*)d_in[2];
  const float* t_emb   = (const float*)d_in[3];
  const float* ff_emb  = (const float*)d_in[4];
  const float* Wi      = (const float*)d_in[5];
  const float* bi      = (const float*)d_in[6];
  const float* Wo      = (const float*)d_in[7];
  const float* bo      = (const float*)d_in[8];
  const float* ln_g    = (const float*)d_in[9];
  const float* ln_b    = (const float*)d_in[10];
  const float* W_cls   = (const float*)d_in[11];
  const float* b_cls   = (const float*)d_in[12];

  char* ws = (char*)d_ws;
  size_t off = 0;
  const size_t ACT = (size_t)MROWS * EMB * sizeof(float);
  float* w_embed  = (float*)(ws + off); off += ACT;
  float* h        = (float*)(ws + off); off += ACT;
  float* qkv      = (float*)(ws + off); off += (size_t)MROWS * N3E * sizeof(float);
  float* attn_out = (float*)(ws + off); off += ACT;
  float* proj     = (float*)(ws + off); off += ACT;
  float* a1       = (float*)(ws + off); off += (size_t)BATCH * SEQ * sizeof(float);
  float* a2       = (float*)(ws + off); off += (size_t)BATCH * SEQ * sizeof(float);
  _Float16* hf    = (_Float16*)(ws + off); off += (size_t)MROWS * KP * 2;
  _Float16* af    = (_Float16*)(ws + off); off += (size_t)MROWS * KP * 2;
  _Float16* qf    = (_Float16*)(ws + off); off += (size_t)BATCH * HEADS * SEQ * DPAD * 2;
  _Float16* kf    = (_Float16*)(ws + off); off += (size_t)BATCH * HEADS * SEQ * DPAD * 2;
  _Float16* vtf   = (_Float16*)(ws + off); off += (size_t)BATCH * HEADS * SEQ * DPAD * 2;
  _Float16* Wif   = (_Float16*)(ws + off); off += (size_t)WIF_ROWS * KP * 2;
  _Float16* Wof   = (_Float16*)(ws + off); off += (size_t)WOF_ROWS * KP * 2;

  tsa_embed_kernel<<<(MROWS * EMB + 255) / 256, 256, 0, stream>>>(
      tokens, W_emb, pos_emb, w_embed, h);

  const long ACTP  = (long)MROWS * KP;
  const long QKVP  = (long)BATCH * HEADS * SEQ * DPAD;

  for (int l = 0; l < NLAY; ++l) {
    // weights -> padded f16
    tsa_cvt_pad_kernel<<<(WIF_ROWS * KP + 255) / 256, 256, 0, stream>>>(
        Wi + (long)l * N3E * EMB, Wif, N3E, (long)WIF_ROWS * KP);
    tsa_cvt_pad_kernel<<<(WOF_ROWS * KP + 255) / 256, 256, 0, stream>>>(
        Wo + (long)l * EMB * EMB, Wof, EMB, (long)WOF_ROWS * KP);
    // h -> padded f16
    tsa_cvt_pad_kernel<<<(int)((ACTP + 255) / 256), 256, 0, stream>>>(
        h, hf, MROWS, ACTP);
    // QKV projection: (M,900) = hf @ Wif^T + bi
    tsa_gemm_f16_kernel<<<dim3(MROWS / 16, 8), 256, 0, stream>>>(
        hf, Wif, bi + (long)l * N3E, qkv, N3E);
    // qkv -> head-major padded f16 (+ transposed V)
    tsa_cvt_qkv_kernel<<<(int)((QKVP + 255) / 256), 256, 0, stream>>>(
        qkv, qf, kf, vtf);
    // attention (64KB V^T + 32KB scores dynamic LDS)
    tsa_attention_kernel<<<BATCH * HEADS * (SEQ / 16), 32, 96 * 1024, stream>>>(
        qf, kf, vtf, attn_out);
    // out projection
    tsa_cvt_pad_kernel<<<(int)((ACTP + 255) / 256), 256, 0, stream>>>(
        attn_out, af, MROWS, ACTP);
    tsa_gemm_f16_kernel<<<dim3(MROWS / 16, 3), 256, 0, stream>>>(
        af, Wof, bo + (long)l * EMB, proj, EMB);
    // task attentions
    tsa_task_attn_kernel<<<BATCH, SEQ, 0, stream>>>(
        w_embed, t_emb + (long)l * EMB, 2u * l + 0x2545F491u, a1);
    tsa_task_attn_kernel<<<BATCH, SEQ, 0, stream>>>(
        h, ff_emb + (long)l * EMB, 2u * l + 1u + 0x2545F491u, a2);
    // h = LN(proj + w_embed*a1 + h*a2)
    tsa_fuse_ln_kernel<<<MROWS / 4, 128, 0, stream>>>(
        proj, w_embed, h, a1, a2, ln_g + (long)l * EMB, ln_b + (long)l * EMB, h);
  }

  tsa_classifier_kernel<<<1, 64, 0, stream>>>(h, W_cls, b_cls, (float*)d_out);
}
